// Attention_40596030882545
// MI455X (gfx1250) — matrix operand order repaired and checked by reading, not compiled, not run on hardware
//
#include <hip/hip_runtime.h>
#include <hip/hip_bf16.h>
#include <math.h>

// Problem constants (match the reference)
#define BB   2048
#define TT   200
#define DD   64
#define FF   256     // 4*D concat features
#define HH1  128
#define HH2  64
#define NTILE 13     // ceil(208/16), T padded to 208
#define TPAD  208
#define NEG_INF_F (-4294967295.0f)   // -2^32 + 1

typedef __attribute__((ext_vector_type(16))) _Float16 v16h;
typedef __attribute__((ext_vector_type(8)))  _Float16 v8h;
typedef __attribute__((ext_vector_type(8)))  float    v8f;

// ---- A fragment for V_WMMA_F32_16X16X32_F16 from row-major f16 LDS [16][ldE].
// ISA 7.12.2 (16-bit A 16x32): M = lane&15 for both halves; lanes 0-15 hold
// K = {0..7, 16..23}, lanes 16-31 hold K = {8..15, 24..31}.  Per lane that is
// two contiguous 8-half (16 B) runs -> two ds_load_b128.
__device__ inline v16h load_a_frag(const _Float16* E, int ldE, int K0, int lane) {
  const int m    = lane & 15;
  const int koff = K0 + ((lane & 16) ? 8 : 0);
  const _Float16* p = E + m * ldE + koff;
  const v8h lo = *(const v8h*)p;          // K = koff .. koff+7
  const v8h hi = *(const v8h*)(p + 16);   // K = koff+16 .. koff+23
  return __builtin_shufflevector(lo, hi, 0, 1, 2, 3, 4, 5, 6, 7,
                                         8, 9, 10, 11, 12, 13, 14, 15);
}

// ---- B fragment from pre-swizzled LDS weights: frag-major layout
// Wf[(frag*32 + lane)*16 + j], where slot j holds W[K0 + (j&8?16:0) + (lane&16?8:0) + (j&7),
// N0 + (lane&15)].  Two aligned ds_load_b128 per fragment, zero packing VALU.
__device__ inline v16h load_b_frag(const _Float16* Wf, int frag, int lane) {
  const _Float16* p = Wf + ((size_t)frag * 32 + lane) * 16;
  const v8h lo = *(const v8h*)p;
  const v8h hi = *(const v8h*)(p + 8);
  return __builtin_shufflevector(lo, hi, 0, 1, 2, 3, 4, 5, 6, 7,
                                         8, 9, 10, 11, 12, 13, 14, 15);
}

__global__ __launch_bounds__(128) void attn_scorer_kernel(
    const float* __restrict__ query,   // (B, 64)
    const float* __restrict__ key,     // (B, 200, 64)
    const float* __restrict__ value,   // (B, 200, 64)
    const int*   __restrict__ mask,    // (B, 200)
    const float* __restrict__ W1,      // (256, 128)
    const float* __restrict__ b1,      // (128)
    const float* __restrict__ W2,      // (128, 64)
    const float* __restrict__ b2,      // (64)
    const float* __restrict__ Wo,      // (64, 1)
    const float* __restrict__ bo,      // (1)
    float*       __restrict__ out)     // (B, 64)
{
  const int b    = blockIdx.x;
  const int tid  = threadIdx.x;
  const int lane = tid & 31;
  const int wave = tid >> 5;

  // ---- LDS (~147 KB) ----
  alignas(16) __shared__ _Float16 sW1f[FF * HH1];     // 64 KB, B-frag layout (64 frags)
  alignas(16) __shared__ _Float16 sW2f[HH1 * HH2];    // 16 KB, B-frag layout (16 frags)
  __shared__ float    sQ[DD];
  __shared__ float    sB1[HH1];
  __shared__ float    sB2[HH2];
  __shared__ float    sWo[HH2];
  alignas(16) __shared__ _Float16 sEmb[4][16 * FF];   // 32 KB, per-wave emb tile
  alignas(16) __shared__ _Float16 sH1 [4][16 * HH1];  // 16 KB, per-wave hidden1
  __shared__ float    sH2 [4][16 * HH2];              // 16 KB, per-wave hidden2
  __shared__ float    sScore[TPAD];
  __shared__ float    sRed[128];

  const size_t bT   = (size_t)b * TT;
  const float* keyB = key   + bT * DD;
  const float* valB = value + bT * DD;
  const int*   mskB = mask  + bT;

  // Stage W1 (f32 -> f16) directly into B-fragment layout.
  // idx = ((kb*8 + nt)*32 + L)*16 + j  ->  W1[k][n], k = kb*32 + (j&8?16:0) + (L&16?8:0) + (j&7),
  // n = nt*16 + (L&15).  Strided global reads, but W1 stays resident in L2.
  for (int idx = tid; idx < FF * HH1; idx += 128) {
    const int j  = idx & 15;
    const int L  = (idx >> 4) & 31;
    const int f  = idx >> 9;          // frag id 0..63
    const int kb = f >> 3;
    const int nt = f & 7;
    const int k  = kb * 32 + ((j & 8) ? 16 : 0) + ((L & 16) ? 8 : 0) + (j & 7);
    const int n  = nt * 16 + (L & 15);
    sW1f[idx] = (_Float16)W1[k * HH1 + n];
  }
  // Same for W2 (16 frags: kb 0..3, nt 0..3).
  for (int idx = tid; idx < HH1 * HH2; idx += 128) {
    const int j  = idx & 15;
    const int L  = (idx >> 4) & 31;
    const int f  = idx >> 9;          // frag id 0..15
    const int kb = f >> 2;
    const int nt = f & 3;
    const int k  = kb * 32 + ((j & 8) ? 16 : 0) + ((L & 16) ? 8 : 0) + (j & 7);
    const int n  = nt * 16 + (L & 15);
    sW2f[idx] = (_Float16)W2[k * HH2 + n];
  }
  if (tid < DD) {
    sQ[tid]  = query[(size_t)b * DD + tid];
    sWo[tid] = Wo[tid];
    sB2[tid] = b2[tid];
  }
  sB1[tid & 127] = b1[tid & 127];
  if (tid < TPAD - TT) sScore[TT + tid] = -INFINITY;   // padding rows -> exp()=0
  for (int i = tid; i < TT; i += 128)
    __builtin_prefetch(valB + (size_t)i * DD, 0, 1);    // global_prefetch_b8
  __syncthreads();

  const float boV  = bo[0];
  const int   moff = (lane & 16) ? 8 : 0;  // C-fragment M offset for this lane half
  const int   ncol = lane & 15;            // C-fragment N within tile

  // Each wave owns M-tiles of 16 timesteps, round-robin.
  for (int tile = wave; tile < NTILE; tile += 4) {
    _Float16* E = sEmb[wave];

    // Build emb tile [16 x 256] = [q | k | q-k | q*k] in f16 (row-major).
    for (int i = lane; i < 16 * DD; i += 32) {
      const int row = i >> 6;
      const int d   = i & 63;
      const int t   = tile * 16 + row;
      const float kv = (t < TT) ? keyB[(size_t)t * DD + d] : 0.0f;
      const float qv = sQ[d];
      E[row * FF + d]          = (_Float16)qv;
      E[row * FF + DD + d]     = (_Float16)kv;
      E[row * FF + 2 * DD + d] = (_Float16)(qv - kv);
      E[row * FF + 3 * DD + d] = (_Float16)(qv * kv);
    }
    __builtin_amdgcn_wave_barrier();   // order LDS stores before fragment loads

    // ---- GEMM1: (16x256) @ (256x128) + b1, f16 in / f32 accum ----
    v8f acc[8];
#pragma unroll
    for (int nt = 0; nt < 8; ++nt) {
      const float bv = sB1[nt * 16 + ncol];
#pragma unroll
      for (int r = 0; r < 8; ++r) acc[nt][r] = bv;
    }
    for (int kb = 0; kb < 8; ++kb) {
      const v16h a = load_a_frag(E, FF, kb * 32, lane);
#pragma unroll
      for (int nt = 0; nt < 8; ++nt) {
        const v16h bf = load_b_frag(sW1f, kb * 8 + nt, lane);
        acc[nt] = __builtin_amdgcn_wmma_f32_16x16x32_f16(
            false, a, false, bf, (short)0, acc[nt], false, false);
      }
    }

    // ReLU -> f16 staging for GEMM2's A matrix (row-major [16][128]).
    _Float16* H1t = sH1[wave];
#pragma unroll
    for (int nt = 0; nt < 8; ++nt)
#pragma unroll
      for (int r = 0; r < 8; ++r) {
        float v = acc[nt][r];
        v = v > 0.0f ? v : 0.0f;
        H1t[(moff + r) * HH1 + nt * 16 + ncol] = (_Float16)v;
      }
    __builtin_amdgcn_wave_barrier();

    // ---- GEMM2: (16x128) @ (128x64) + b2 ----
    v8f acc2[4];
#pragma unroll
    for (int nt = 0; nt < 4; ++nt) {
      const float bv = sB2[nt * 16 + ncol];
#pragma unroll
      for (int r = 0; r < 8; ++r) acc2[nt][r] = bv;
    }
    for (int kb = 0; kb < 4; ++kb) {
      const v16h a2 = load_a_frag(H1t, HH1, kb * 32, lane);
#pragma unroll
      for (int nt = 0; nt < 4; ++nt) {
        const v16h bf = load_b_frag(sW2f, kb * 4 + nt, lane);
        acc2[nt] = __builtin_amdgcn_wmma_f32_16x16x32_f16(
            false, a2, false, bf, (short)0, acc2[nt], false, false);
      }
    }

    // ReLU -> f32 staging for the score dot.
    float* H2t = sH2[wave];
#pragma unroll
    for (int nt = 0; nt < 4; ++nt)
#pragma unroll
      for (int r = 0; r < 8; ++r) {
        float v = acc2[nt][r];
        H2t[(moff + r) * HH2 + nt * 16 + ncol] = v > 0.0f ? v : 0.0f;
      }
    __builtin_amdgcn_wave_barrier();

    // Score: h2 . Wo + bo, then mask (reference's NEG_INF for masked rows).
    if (lane < 16) {
      const int t = tile * 16 + lane;
      if (t < TT) {
        float s = boV;
        const float* hrow = H2t + lane * HH2;
#pragma unroll 8
        for (int d = 0; d < HH2; ++d) s += hrow[d] * sWo[d];
        sScore[t] = (mskB[t] == 0) ? NEG_INF_F : s;
      }
    }
  }
  __syncthreads();

  // ---- Softmax over T (block reduction) ----
  float mloc = -INFINITY;
  for (int i = tid; i < TPAD; i += 128) mloc = fmaxf(mloc, sScore[i]);
  sRed[tid] = mloc;
  __syncthreads();
  for (int s = 64; s > 0; s >>= 1) {
    if (tid < s) sRed[tid] = fmaxf(sRed[tid], sRed[tid + s]);
    __syncthreads();
  }
  const float mx = sRed[0];
  __syncthreads();

  float sloc = 0.0f;
  for (int i = tid; i < TPAD; i += 128) {
    const float e = __expf(sScore[i] - mx);   // padding rows -> 0
    sScore[i] = e;
    sloc += e;
  }
  sRed[tid] = sloc;
  __syncthreads();
  for (int s = 64; s > 0; s >>= 1) {
    if (tid < s) sRed[tid] += sRed[tid + s];
    __syncthreads();
  }
  const float inv = 1.0f / sRed[0];
  __syncthreads();

  // ---- out[d] = (sum_t e_t * value[t,d]) * inv ; coalesced over d ----
  const int g = tid >> 6;       // two groups split the T range
  const int d = tid & 63;
  float accv = 0.0f;
  for (int t = g * 100; t < g * 100 + 100; ++t)
    accv += sScore[t] * valB[(size_t)t * DD + d];
  sRed[tid] = accv;
  __syncthreads();
  if (tid < 64)
    out[(size_t)b * DD + tid] = (sRed[tid] + sRed[tid + 64]) * inv;
}

extern "C" void kernel_launch(void* const* d_in, const int* in_sizes, int n_in,
                              void* d_out, int out_size, void* d_ws, size_t ws_size,
                              hipStream_t stream) {
  (void)n_in; (void)d_ws; (void)ws_size; (void)out_size;
  const float* query = (const float*)d_in[0];
  const float* key   = (const float*)d_in[1];
  const float* value = (const float*)d_in[2];
  const int*   mask  = (const int*)  d_in[3];
  const float* W1    = (const float*)d_in[4];
  const float* b1    = (const float*)d_in[5];
  const float* W2    = (const float*)d_in[6];
  const float* b2    = (const float*)d_in[7];
  const float* Wo    = (const float*)d_in[8];
  const float* bo    = (const float*)d_in[9];
  float*       out   = (float*)d_out;

  const int B = in_sizes[0] / DD;   // 2048
  attn_scorer_kernel<<<B, 128, 0, stream>>>(query, key, value, mask,
                                            W1, b1, W2, b2, Wo, bo, out);
}
